// VanillaSequenceToSequence_27264452395705
// MI455X (gfx1250) — matrix operand order repaired and checked by reading, not compiled, not run on hardware
//
#include <hip/hip_runtime.h>
#include <hip/hip_bf16.h>
#include <math.h>

// ---------------------------------------------------------------------------
// GRU seq2seq for MI455X (gfx1250, wave32, WMMA).
// - All weights converted fp32 -> bf16 once into d_ws: 4 x 6.3MB GRU mats +
//   65.5MB out_W ~= 91MB, resident in 192MB L2 -> steady-state GEMMs run out
//   of L2, never HBM (23.3 TB/s is irrelevant after the first touch).
// - v_wmma_f32_16x16x32_bf16, f32 accumulation. Each wave owns a 16x32
//   output slab (two N-tiles, shared A operand -> 3 loads/WMMA), and the
//   per-step GRU kernel fuses the x@Wih^T and h@Whh^T GEMMs (4 WMMAs per
//   K-chunk) to halve the serial per-timestep critical path.
// - A/B registers, in contiguous k order (k = 16*(lane/16)+s), are filled with direct 16B global
//   loads (weights row-major [N,K] -> each lane reads 16 consecutive k).
// ---------------------------------------------------------------------------

typedef __attribute__((ext_vector_type(16))) __bf16 v16bf;
typedef __attribute__((ext_vector_type(8)))  float  v8f;

#define B_  32
#define H_  1024
#define S_  128
#define T_  64
#define V_  32000
#define G_  3072   /* 3*H */

__device__ __forceinline__ unsigned short f2bf(float f) {
  unsigned int u = __float_as_uint(f);
  u += 0x7FFFu + ((u >> 16) & 1u);          // round-to-nearest-even
  return (unsigned short)(u >> 16);
}

union Tile { v16bf v; uint4 u[2]; };

// ---------------- elementwise / setup kernels ------------------------------

__global__ void k_f32_to_bf16(const float* __restrict__ src,
                              unsigned short* __restrict__ dst, size_t n) {
  size_t i = (size_t)blockIdx.x * blockDim.x + threadIdx.x;
  if (i < n) dst[i] = f2bf(src[i]);
}

__global__ void k_init_h(const float* __restrict__ h0,
                         float* __restrict__ h,
                         unsigned short* __restrict__ hbf) {
  int i = blockIdx.x * blockDim.x + threadIdx.x;      // B*H threads
  float x = h0[i];
  h[i] = x;
  hbf[i] = f2bf(x);
}

__global__ void k_init_tok(int* __restrict__ tok) {
  int i = threadIdx.x;
  if (i < B_) tok[i] = 1;                             // SOS_IND
}

__global__ void k_enc_embed(const int* __restrict__ input,     // [B,S]
                            const float* __restrict__ emb,     // [V,H] f32
                            unsigned short* __restrict__ xbf)  // [S,B,H] bf16
{
  size_t idx = (size_t)blockIdx.x * blockDim.x + threadIdx.x;  // S*B*H
  int j = (int)(idx & (H_ - 1));
  int b = (int)((idx >> 10) & (B_ - 1));
  int t = (int)(idx >> 15);
  int tok = input[b * S_ + t];
  xbf[idx] = f2bf(emb[(size_t)tok * H_ + j]);
}

__global__ void k_dec_embed(const int* __restrict__ tok,       // [B]
                            const float* __restrict__ emb,     // [V,H] f32
                            unsigned short* __restrict__ xbf)  // [B,H] bf16
{
  int idx = blockIdx.x * blockDim.x + threadIdx.x;    // B*H threads
  int b = idx >> 10, j = idx & (H_ - 1);
  xbf[idx] = f2bf(emb[(size_t)tok[b] * H_ + j]);
}

// ---------------- fused GRU dual-GEMM --------------------------------------
// gi[m,n] = sum_k X[m,k]*Wih[n,k];  gh[m,n] = sum_k Hst[m,k]*Whh[n,k].
// One wave owns a 16x32 slab of BOTH outputs: 4 accumulators, 4 WMMAs per
// K-chunk (12 x 16B loads). 192 waves total -> 24 blocks of 256.
__global__ void __launch_bounds__(256)
k_gru_gemm(const unsigned short* __restrict__ X,    // [32,H] bf16 (x_t)
           const unsigned short* __restrict__ Hst,  // [32,H] bf16 (h_{t-1})
           const unsigned short* __restrict__ Wih,  // [3H,H] bf16
           const unsigned short* __restrict__ Whh,  // [3H,H] bf16
           float* __restrict__ gi, float* __restrict__ gh)
{
  int wave = (blockIdx.x << 3) | (threadIdx.x >> 5);
  int lane = threadIdx.x & 31;
  int mt = wave & 1;                  // 2 M-tiles (B=32)
  int np = wave >> 1;                 // N-pair: 96 pairs cover 3H=3072
  if (np >= G_ / 32) return;          // wave-uniform

  int hi = lane >> 4;
  int rA = (mt << 4) + (lane & 15);
  int n0 = (np << 5) + (lane & 15);
  int n1 = n0 + 16;
  const unsigned short* xp  = X   + (size_t)rA * H_ + (hi ? 16 : 0);
  const unsigned short* hp  = Hst + (size_t)rA * H_ + (hi ? 16 : 0);
  const unsigned short* wi0 = Wih + (size_t)n0 * H_ + (hi ? 16 : 0);
  const unsigned short* wi1 = Wih + (size_t)n1 * H_ + (hi ? 16 : 0);
  const unsigned short* wh0 = Whh + (size_t)n0 * H_ + (hi ? 16 : 0);
  const unsigned short* wh1 = Whh + (size_t)n1 * H_ + (hi ? 16 : 0);

  v8f ai0 = {}, ai1 = {}, ah0 = {}, ah1 = {};
#pragma unroll 2
  for (int k0 = 0; k0 < H_; k0 += 32) {
    Tile ax, ah, bi0, bi1, bh0, bh1;
    ax.u[0]  = *(const uint4*)(xp  + k0);
    ax.u[1]  = *(const uint4*)(xp  + k0 + 8);
    ah.u[0]  = *(const uint4*)(hp  + k0);
    ah.u[1]  = *(const uint4*)(hp  + k0 + 8);
    bi0.u[0] = *(const uint4*)(wi0 + k0);
    bi0.u[1] = *(const uint4*)(wi0 + k0 + 8);
    bi1.u[0] = *(const uint4*)(wi1 + k0);
    bi1.u[1] = *(const uint4*)(wi1 + k0 + 8);
    bh0.u[0] = *(const uint4*)(wh0 + k0);
    bh0.u[1] = *(const uint4*)(wh0 + k0 + 8);
    bh1.u[0] = *(const uint4*)(wh1 + k0);
    bh1.u[1] = *(const uint4*)(wh1 + k0 + 8);
    ai0 = __builtin_amdgcn_wmma_f32_16x16x32_bf16(false, ax.v, false, bi0.v,
                                                  (short)0, ai0, false, false);
    ai1 = __builtin_amdgcn_wmma_f32_16x16x32_bf16(false, ax.v, false, bi1.v,
                                                  (short)0, ai1, false, false);
    ah0 = __builtin_amdgcn_wmma_f32_16x16x32_bf16(false, ah.v, false, bh0.v,
                                                  (short)0, ah0, false, false);
    ah1 = __builtin_amdgcn_wmma_f32_16x16x32_bf16(false, ah.v, false, bh1.v,
                                                  (short)0, ah1, false, false);
  }

  int mbase = (mt << 4) + (hi ? 8 : 0);
#pragma unroll
  for (int j = 0; j < 8; ++j) {
    size_t r = (size_t)(mbase + j) * G_;
    gi[r + n0] = ai0[j];
    gi[r + n1] = ai1[j];
    gh[r + n0] = ah0[j];
    gh[r + n1] = ah1[j];
  }
}

// ---------------- generic WMMA GEMM, 2 N-tiles / wave ----------------------
// Out[m,n] = sum_k X[m,k]*W[n,k] + bias[n].  Used for the vocab projection:
// 2000 N-tiles -> 1000 pairs -> 2000 waves -> 250 blocks.
__global__ void __launch_bounds__(256)
k_wmma_gemm2(const unsigned short* __restrict__ X,   // [32,K] bf16
             const unsigned short* __restrict__ W,   // [N,K] bf16
             const float* __restrict__ bias,         // [N] or null
             float* __restrict__ Out,
             int K, int NtilePairs, size_t ldOut)
{
  int wave = (blockIdx.x << 3) | (threadIdx.x >> 5);
  int lane = threadIdx.x & 31;
  int mt = wave & 1;
  int np = wave >> 1;
  if (np >= NtilePairs) return;       // wave-uniform

  int hi = lane >> 4;
  int rA = (mt << 4) + (lane & 15);
  int n0 = (np << 5) + (lane & 15);
  int n1 = n0 + 16;
  const unsigned short* xp  = X + (size_t)rA * K + (hi ? 16 : 0);
  const unsigned short* wp0 = W + (size_t)n0 * K + (hi ? 16 : 0);
  const unsigned short* wp1 = W + (size_t)n1 * K + (hi ? 16 : 0);

  v8f acc0 = {}, acc1 = {};
#pragma unroll 2
  for (int k0 = 0; k0 < K; k0 += 32) {
    Tile a, b0, b1;
    a.u[0]  = *(const uint4*)(xp  + k0);
    a.u[1]  = *(const uint4*)(xp  + k0 + 8);
    b0.u[0] = *(const uint4*)(wp0 + k0);
    b0.u[1] = *(const uint4*)(wp0 + k0 + 8);
    b1.u[0] = *(const uint4*)(wp1 + k0);
    b1.u[1] = *(const uint4*)(wp1 + k0 + 8);
    acc0 = __builtin_amdgcn_wmma_f32_16x16x32_bf16(false, a.v, false, b0.v,
                                                   (short)0, acc0, false, false);
    acc1 = __builtin_amdgcn_wmma_f32_16x16x32_bf16(false, a.v, false, b1.v,
                                                   (short)0, acc1, false, false);
  }

  float bv0 = bias ? bias[n0] : 0.0f;
  float bv1 = bias ? bias[n1] : 0.0f;
  int mbase = (mt << 4) + (hi ? 8 : 0);
#pragma unroll
  for (int j = 0; j < 8; ++j) {
    size_t r = (size_t)(mbase + j) * ldOut;
    Out[r + n0] = acc0[j] + bv0;
    Out[r + n1] = acc1[j] + bv1;
  }
}

// ---------------- GRU gate fusion ------------------------------------------

__global__ void k_gru_gate(const float* __restrict__ gi,   // [B,3H]
                           const float* __restrict__ gh,   // [B,3H]
                           const float* __restrict__ b_ih, // [3H]
                           const float* __restrict__ b_hh, // [3H]
                           float* __restrict__ h,          // [B,H] in/out
                           unsigned short* __restrict__ hbf)
{
  int idx = blockIdx.x * blockDim.x + threadIdx.x;   // B*H threads
  int b = idx >> 10, j = idx & (H_ - 1);
  const float* gib = gi + (size_t)b * G_;
  const float* ghb = gh + (size_t)b * G_;
  float ir  = gib[j]           + b_ih[j];
  float iz  = gib[H_ + j]      + b_ih[H_ + j];
  float in_ = gib[2 * H_ + j]  + b_ih[2 * H_ + j];
  float hr  = ghb[j]           + b_hh[j];
  float hz  = ghb[H_ + j]      + b_hh[H_ + j];
  float hn  = ghb[2 * H_ + j]  + b_hh[2 * H_ + j];
  float r = 1.0f / (1.0f + expf(-(ir + hr)));
  float z = 1.0f / (1.0f + expf(-(iz + hz)));
  float nn = tanhf(in_ + r * hn);
  float h2 = (1.0f - z) * nn + z * h[idx];
  h[idx] = h2;
  hbf[idx] = f2bf(h2);
}

// ---------------- argmax over vocab ----------------------------------------

__global__ void k_argmax(const float* __restrict__ logits, size_t rowStride,
                         int* __restrict__ tok,
                         float* __restrict__ tokOut, int tokOutStride)
{
  __shared__ float sval[256];
  __shared__ int   sidx[256];
  int b = blockIdx.x;
  const float* row = logits + (size_t)b * rowStride;
  float best = -INFINITY; int bi = 0;
  for (int v = threadIdx.x; v < V_; v += blockDim.x) {
    float x = row[v];
    if (x > best) { best = x; bi = v; }
  }
  sval[threadIdx.x] = best; sidx[threadIdx.x] = bi;
  __syncthreads();
  for (int s = 128; s > 0; s >>= 1) {
    if ((int)threadIdx.x < s) {
      float ov = sval[threadIdx.x + s]; int oi = sidx[threadIdx.x + s];
      if (ov > sval[threadIdx.x] ||
          (ov == sval[threadIdx.x] && oi < sidx[threadIdx.x])) {
        sval[threadIdx.x] = ov; sidx[threadIdx.x] = oi;
      }
    }
    __syncthreads();
  }
  if (threadIdx.x == 0) {
    tok[b] = sidx[0];
    tokOut[(size_t)b * tokOutStride] = (float)sidx[0];
  }
}

__global__ void k_copy_h(const float* __restrict__ h, float* __restrict__ out) {
  int i = blockIdx.x * blockDim.x + threadIdx.x;     // B*H threads
  out[i] = h[i];
}

// ---------------- host-side orchestration ----------------------------------

extern "C" void kernel_launch(void* const* d_in, const int* in_sizes, int n_in,
                              void* d_out, int out_size, void* d_ws, size_t ws_size,
                              hipStream_t stream) {
  (void)in_sizes; (void)n_in; (void)out_size; (void)ws_size;

  const int*   input    = (const int*)  d_in[0];
  const float* hidden   = (const float*)d_in[1];
  const float* enc_emb  = (const float*)d_in[3];
  const float* enc_W_ih = (const float*)d_in[4];
  const float* enc_W_hh = (const float*)d_in[5];
  const float* enc_b_ih = (const float*)d_in[6];
  const float* enc_b_hh = (const float*)d_in[7];
  const float* dec_emb  = (const float*)d_in[8];
  const float* dec_W_ih = (const float*)d_in[9];
  const float* dec_W_hh = (const float*)d_in[10];
  const float* dec_b_ih = (const float*)d_in[11];
  const float* dec_b_hh = (const float*)d_in[12];
  const float* out_W    = (const float*)d_in[13];
  const float* out_b    = (const float*)d_in[14];

  // ---- carve workspace (256B aligned) ----
  char* p = (char*)d_ws;
  auto carve = [&](size_t bytes) -> void* {
    void* r = (void*)p;
    p += (bytes + 255) & ~(size_t)255;
    return r;
  };
  unsigned short* encWih = (unsigned short*)carve((size_t)G_ * H_ * 2);
  unsigned short* encWhh = (unsigned short*)carve((size_t)G_ * H_ * 2);
  unsigned short* decWih = (unsigned short*)carve((size_t)G_ * H_ * 2);
  unsigned short* decWhh = (unsigned short*)carve((size_t)G_ * H_ * 2);
  unsigned short* outWbf = (unsigned short*)carve((size_t)V_ * H_ * 2);
  unsigned short* xenc   = (unsigned short*)carve((size_t)S_ * B_ * H_ * 2);
  unsigned short* xdec   = (unsigned short*)carve((size_t)B_ * H_ * 2);
  float*          hbuf   = (float*)         carve((size_t)B_ * H_ * 4);
  unsigned short* hbf    = (unsigned short*)carve((size_t)B_ * H_ * 2);
  float*          gi     = (float*)         carve((size_t)B_ * G_ * 4);
  float*          gh     = (float*)         carve((size_t)B_ * G_ * 4);
  int*            tok    = (int*)           carve((size_t)B_ * 4);

  float* out       = (float*)d_out;
  float* outToks   = out;                           // [B,T] (written as float)
  float* outHidden = out + (size_t)B_ * T_;         // [1,B,H]
  float* outLogits = outHidden + (size_t)B_ * H_;   // [B,T,V]

  // ---- one-time fp32 -> bf16 weight conversion (L2-resident afterwards) ---
  {
    size_t n = (size_t)G_ * H_;
    int blk = 256; int grd = (int)((n + blk - 1) / blk);
    k_f32_to_bf16<<<grd, blk, 0, stream>>>(enc_W_ih, encWih, n);
    k_f32_to_bf16<<<grd, blk, 0, stream>>>(enc_W_hh, encWhh, n);
    k_f32_to_bf16<<<grd, blk, 0, stream>>>(dec_W_ih, decWih, n);
    k_f32_to_bf16<<<grd, blk, 0, stream>>>(dec_W_hh, decWhh, n);
    size_t nv = (size_t)V_ * H_;
    k_f32_to_bf16<<<(int)((nv + blk - 1) / blk), blk, 0, stream>>>(out_W, outWbf, nv);
  }

  k_init_h<<<(B_ * H_) / 256, 256, 0, stream>>>(hidden, hbuf, hbf);
  k_enc_embed<<<(S_ * B_ * H_) / 256, 256, 0, stream>>>(input, enc_emb, xenc);

  // ---- encoder: 128 sequential GRU steps (2 kernels per step) ----
  const int gruBlocks = (2 * (G_ / 32) * 32) / 256;   // 192 waves -> 24 blocks
  for (int t = 0; t < S_; ++t) {
    const unsigned short* xt = xenc + (size_t)t * B_ * H_;
    k_gru_gemm<<<gruBlocks, 256, 0, stream>>>(xt, hbf, encWih, encWhh, gi, gh);
    k_gru_gate<<<(B_ * H_) / 256, 256, 0, stream>>>(gi, gh, enc_b_ih, enc_b_hh,
                                                    hbuf, hbf);
  }

  // ---- decoder: 64 sequential greedy steps ----
  k_init_tok<<<1, 32, 0, stream>>>(tok);
  const int vocabBlocks = (2 * (V_ / 32) * 32) / 256; // 2000 waves -> 250 blocks
  for (int t = 0; t < T_; ++t) {
    k_dec_embed<<<(B_ * H_) / 256, 256, 0, stream>>>(tok, dec_emb, xdec);
    k_gru_gemm<<<gruBlocks, 256, 0, stream>>>(xdec, hbf, decWih, decWhh, gi, gh);
    k_gru_gate<<<(B_ * H_) / 256, 256, 0, stream>>>(gi, gh, dec_b_ih, dec_b_hh,
                                                    hbuf, hbf);
    // logits -> directly into d_out [B,T,V] slice for this t
    float* lt = outLogits + (size_t)t * V_;
    k_wmma_gemm2<<<vocabBlocks, 256, 0, stream>>>(hbf, outWbf, out_b, lt,
                                                  H_, V_ / 32,
                                                  (size_t)T_ * V_);
    k_argmax<<<B_, 256, 0, stream>>>(lt, (size_t)T_ * V_, tok,
                                     outToks + t, T_);
  }

  k_copy_h<<<(B_ * H_) / 256, 256, 0, stream>>>(hbuf, outHidden);
}
